// Toy_MultiHeadAttention_13254269075733
// MI455X (gfx1250) — matrix operand order repaired and checked by reading, not compiled, not run on hardware
//
#include <hip/hip_runtime.h>
#include <hip/hip_bf16.h>

// ---------------- problem constants ----------------
constexpr int Bc  = 4;
constexpr int Lc  = 2048;
constexpr int Dc  = 1024;
constexpr int Hc  = 16;
constexpr int HDc = 64;
constexpr int D3c = 3 * Dc;          // 3072
constexpr int BLc = Bc * Lc;         // 8192

typedef __attribute__((ext_vector_type(16))) __bf16 v16bf;
typedef __attribute__((ext_vector_type(8)))  __bf16 v8bf;
typedef __attribute__((ext_vector_type(8)))  float  v8f;

// float -> bf16 (round-to-nearest-even) via bit ops
static __device__ __forceinline__ __bf16 f2bf(float f) {
    unsigned u = __builtin_bit_cast(unsigned, f);
    unsigned r = u + 0x7FFFu + ((u >> 16) & 1u);
    unsigned short h = (unsigned short)(r >> 16);
    return __builtin_bit_cast(__bf16, h);
}

// ---------------- CDNA5 async LDS copy (ASYNCcnt path) ----------------
// GLOBAL_LOAD_ASYNC_TO_LDS_B128: per-lane 16B copy global->LDS, no VGPR data,
// tracked with ASYNCcnt (CDNA5 ISA).
static __device__ __forceinline__ void async_copy_b128(void* lds_dst, const void* gsrc) {
    unsigned ldsa = (unsigned)(reinterpret_cast<uintptr_t>(lds_dst)); // low 32 bits = LDS offset
    asm volatile("global_load_async_to_lds_b128 %0, %1, off"
                 :: "v"(ldsa), "v"(gsrc) : "memory");
}
static __device__ __forceinline__ void wait_async0() {
    asm volatile("s_wait_asynccnt 0x0" ::: "memory");
}

// ---- WMMA fragment loaders (bf16 16x16x32), work on global or LDS pointers ----
// A (16x32, MxK): lane&15 = row; (lane>>4)*8 = k-chunk; elems 0-7 = K k0..k0+7, 8-15 = K k0+16..k0+23
static __device__ __forceinline__ v16bf load_a_frag(const __bf16* base, int ld) {
    const int lane = threadIdx.x & 31;
    const __bf16* p = base + (lane & 15) * (long)ld + ((lane >> 4) * 8);
    v8bf lo = *(const v8bf*)(p);
    v8bf hi = *(const v8bf*)(p + 16);
    return __builtin_shufflevector(lo, hi, 0,1,2,3,4,5,6,7,8,9,10,11,12,13,14,15);
}
// B (32x16, KxN) from Bt[N][K]: lane&15 = column; (lane>>4)*8 = k-chunk; elems 0-7 = K k0..k0+7, 8-15 = K k0+16..k0+23
static __device__ __forceinline__ v16bf load_b_frag(const __bf16* base, int ld) {
    const int lane = threadIdx.x & 31;
    const __bf16* p = base + (lane & 15) * (long)ld + ((lane >> 4) * 8);
    v8bf lo = *(const v8bf*)(p);
    v8bf hi = *(const v8bf*)(p + 16);
    return __builtin_shufflevector(lo, hi, 0,1,2,3,4,5,6,7,8,9,10,11,12,13,14,15);
}

// ---------------- conversion kernels ----------------
__global__ void cvt_f32_bf16_kernel(const float* __restrict__ in, __bf16* __restrict__ out, int n) {
    int i = blockIdx.x * blockDim.x + threadIdx.x;
    if (i < n) out[i] = f2bf(in[i]);
}

// W[K][N] (fp32) -> Wt[N][K] (bf16)
__global__ void transpose_cvt_kernel(const float* __restrict__ W, __bf16* __restrict__ Wt, int K, int N) {
    int i = blockIdx.x * blockDim.x + threadIdx.x;
    if (i < N * K) {
        int n = i / K, k = i - n * K;
        Wt[i] = f2bf(W[(long)k * N + n]);
    }
}

// Vt[(b*H+h)*HD + d][l] = qkv[(b*L+l)*3D + 2D + h*HD + d]
__global__ void vtrans_kernel(const __bf16* __restrict__ qkv, __bf16* __restrict__ Vt) {
    int i = blockIdx.x * blockDim.x + threadIdx.x;
    if (i < Bc * Hc * HDc * Lc) {
        int l  = i % Lc;
        int t  = i / Lc;
        int d  = t % HDc;
        int bh = t / HDc;
        int h  = bh % Hc, b = bh / Hc;
        Vt[i] = qkv[((long)(b * Lc + l)) * D3c + 2 * Dc + h * HDc + d];
    }
}

// ---------------- async-pipelined bf16 WMMA GEMM ----------------
// C[M][N] = A[M][K] * Bt[N][K]^T ; block = 256 thr = 8 waves (2 M x 4 N), wave tile 64x32.
// 128x32 A and B tiles double-buffered in LDS via global_load_async_to_lds_b128.
template <bool OUT_BF16>
__global__ __launch_bounds__(256)
void gemm_bf16_kernel(const __bf16* __restrict__ A, int lda,
                      const __bf16* __restrict__ Bt, int ldb,
                      void* __restrict__ Cout, int ldc,
                      const float* __restrict__ bias,
                      int K) {
    __shared__ __align__(32) __bf16 tA[2][128][32];   // 2 x 8 KB
    __shared__ __align__(32) __bf16 tB[2][128][32];   // 2 x 8 KB
    const int lane = threadIdx.x & 31;
    const int wave = threadIdx.x >> 5;
    const int wm = (wave >> 2) * 64;
    const int wn = (wave & 3) * 32;
    const int bm0 = blockIdx.y * 128;
    const int bn0 = blockIdx.x * 128;

    auto stage = [&](int buf, int k) {
#pragma unroll
        for (int c = threadIdx.x; c < 512; c += 256) {      // 512 x 16B chunks per tile
            int row = c >> 2, kp = (c & 3) * 8;
            async_copy_b128(&tA[buf][row][kp], A  + (long)(bm0 + row) * lda + k + kp);
            async_copy_b128(&tB[buf][row][kp], Bt + (long)(bn0 + row) * ldb + k + kp);
        }
    };

    stage(0, 0);
    v8f acc[4][2] = {};
    for (int k = 0; k < K; k += 32) {
        const int buf = (k >> 5) & 1;
        wait_async0();                 // our wave's copies (incl. prefetch) have landed
        __syncthreads();               // everyone's copies visible; prev buf fully consumed
        if (k + 32 < K) stage(buf ^ 1, k + 32);   // overlap DMA with WMMA
        v16bf bfrag[2];
#pragma unroll
        for (int j = 0; j < 2; ++j)
            bfrag[j] = load_b_frag(&tB[buf][wn + j * 16][0], 32);
#pragma unroll
        for (int i = 0; i < 4; ++i) {
            v16bf afrag = load_a_frag(&tA[buf][wm + i * 16][0], 32);
#pragma unroll
            for (int j = 0; j < 2; ++j)
                acc[i][j] = __builtin_amdgcn_wmma_f32_16x16x32_bf16(
                    false, afrag, false, bfrag[j], (short)0, acc[i][j], false, false);
        }
    }
    const int nloc  = lane & 15;
    const int rbase = (lane >> 4) * 8;
#pragma unroll
    for (int i = 0; i < 4; ++i) {
#pragma unroll
        for (int j = 0; j < 2; ++j) {
            int col = bn0 + wn + j * 16 + nloc;
            float bv = 0.f;
            if (!OUT_BF16) bv = bias[col];
#pragma unroll
            for (int r = 0; r < 8; ++r) {
                long row = bm0 + wm + i * 16 + rbase + r;
                float v = acc[i][j][r];
                if (OUT_BF16) ((__bf16*)Cout)[row * ldc + col] = f2bf(v);
                else          ((float*)Cout)[row * ldc + col]  = v + bv;
            }
        }
    }
}

// ---------------- flash attention (causal, online softmax, async K/V tiles) ----------------
// grid = (L/64, B*H), block = 128 (4 waves); wave owns 16 query rows.
// K-tile (64 keys x 64 hd) and V-tile (64 hd x 64 keys) staged once per BLOCK via async
// copies (shared by all 4 waves -> 4x less K/V traffic), double-buffered across key blocks.
__global__ __launch_bounds__(128)
void attn_kernel(const __bf16* __restrict__ qkv, const __bf16* __restrict__ Vt,
                 __bf16* __restrict__ O) {
    __shared__ __align__(32) __bf16 tK[2][64][64];    // 2 x 8 KB  [key][hd]
    __shared__ __align__(32) __bf16 tV[2][64][64];    // 2 x 8 KB  [hd][key]
    __shared__ __align__(32) __bf16 ldsP[4][16][64];  // 8 KB, per-wave P re-layout
    const int lane = threadIdx.x & 31;
    const int wave = threadIdx.x >> 5;
    const int bh   = blockIdx.y;
    const int b    = bh >> 4;
    const int h    = bh & 15;
    const int qblk = blockIdx.x;
    const int q0   = qblk * 64 + wave * 16;

    const int nloc  = lane & 15;
    const int rbase = (lane >> 4) * 8;

    auto stage = [&](int buf, int j) {
        const int kb = j * 64;
#pragma unroll
        for (int c = threadIdx.x; c < 512; c += 128) {      // 512 x 16B chunks per tile
            int row = c >> 3, kp = (c & 7) * 8;
            async_copy_b128(&tK[buf][row][kp],
                            qkv + (long)(b * Lc + kb + row) * D3c + Dc + h * HDc + kp);
            async_copy_b128(&tV[buf][row][kp],
                            Vt + ((long)(b * Hc + h) * HDc + row) * Lc + kb + kp);
        }
    };

    // preload Q fragments (HD=64 -> 2 k-steps), strided rows inside qkv buffer
    const __bf16* Qbase = qkv + (long)(b * Lc + q0) * D3c + h * HDc;
    v16bf qa[2];
#pragma unroll
    for (int kk = 0; kk < 2; ++kk) qa[kk] = load_a_frag(Qbase + kk * 32, D3c);

    v8f   oacc[4] = {};
    float rmax[8], rsum[8];
#pragma unroll
    for (int r = 0; r < 8; ++r) { rmax[r] = -1e30f; rsum[r] = 0.f; }

    stage(0, 0);
    for (int j = 0; j <= qblk; ++j) {
        const int buf = j & 1;
        const int kb  = j * 64;
        wait_async0();
        __syncthreads();
        if (j < qblk) stage(buf ^ 1, j + 1);   // stream next key block during compute

        // ---- S = Q K^T from LDS K-tile ----
        v8f s[4] = {};
#pragma unroll
        for (int kk = 0; kk < 2; ++kk) {
#pragma unroll
            for (int f = 0; f < 4; ++f) {
                v16bf kf = load_b_frag(&tK[buf][f * 16][kk * 32], 64);
                s[f] = __builtin_amdgcn_wmma_f32_16x16x32_bf16(
                    false, qa[kk], false, kf, (short)0, s[f], false, false);
            }
        }
        // ---- scale + causal mask ----
        const bool diag = (j == qblk);
#pragma unroll
        for (int f = 0; f < 4; ++f) {
#pragma unroll
            for (int r = 0; r < 8; ++r) {
                float v = s[f][r] * 0.125f;   // 1/sqrt(64)
                if (diag) {
                    int col = kb + f * 16 + nloc;
                    int row = q0 + rbase + r;
                    if (col > row) v = -1e30f;
                }
                s[f][r] = v;
            }
        }
        // ---- online softmax on C-fragment layout (row lives in 16-lane half) ----
#pragma unroll
        for (int r = 0; r < 8; ++r) {
            float v = fmaxf(fmaxf(s[0][r], s[1][r]), fmaxf(s[2][r], s[3][r]));
            v = fmaxf(v, __shfl_xor(v, 1, 32));
            v = fmaxf(v, __shfl_xor(v, 2, 32));
            v = fmaxf(v, __shfl_xor(v, 4, 32));
            v = fmaxf(v, __shfl_xor(v, 8, 32));
            float nm   = fmaxf(rmax[r], v);
            float corr = __expf(rmax[r] - nm);
            rmax[r] = nm;
            rsum[r] *= corr;
#pragma unroll
            for (int f = 0; f < 4; ++f) oacc[f][r] *= corr;
            float psum = 0.f;
#pragma unroll
            for (int f = 0; f < 4; ++f) {
                float p = __expf(s[f][r] - nm);
                psum += p;
                ldsP[wave][rbase + r][f * 16 + nloc] = f2bf(p);  // C-frag -> row-major
            }
            psum += __shfl_xor(psum, 1, 32);
            psum += __shfl_xor(psum, 2, 32);
            psum += __shfl_xor(psum, 4, 32);
            psum += __shfl_xor(psum, 8, 32);
            rsum[r] += psum;
        }
        // ---- O += P V  (P as A-frag from LDS; V as B-frag from LDS V-tile) ----
#pragma unroll
        for (int kk = 0; kk < 2; ++kk) {
            const __bf16* pp = &ldsP[wave][nloc][kk * 32 + rbase];
            v8bf lo = *(const v8bf*)(pp);
            v8bf hi = *(const v8bf*)(pp + 16);
            v16bf pf = __builtin_shufflevector(lo, hi, 0,1,2,3,4,5,6,7,8,9,10,11,12,13,14,15);
#pragma unroll
            for (int f = 0; f < 4; ++f) {
                v16bf vf = load_b_frag(&tV[buf][f * 16][kk * 32], 64);
                oacc[f] = __builtin_amdgcn_wmma_f32_16x16x32_bf16(
                    false, pf, false, vf, (short)0, oacc[f], false, false);
            }
        }
    }
    // ---- normalize + store O as bf16 [B][L][D] ----
#pragma unroll
    for (int f = 0; f < 4; ++f) {
#pragma unroll
        for (int r = 0; r < 8; ++r) {
            int row = q0 + rbase + r;
            float v = oacc[f][r] / rsum[r];
            O[(long)(b * Lc + row) * Dc + h * HDc + f * 16 + nloc] = f2bf(v);
        }
    }
}

// ---------------- host launcher ----------------
extern "C" void kernel_launch(void* const* d_in, const int* in_sizes, int n_in,
                              void* d_out, int out_size, void* d_ws, size_t ws_size,
                              hipStream_t stream) {
    const float* x    = (const float*)d_in[0];
    const float* Wqkv = (const float*)d_in[1];
    const float* Wout = (const float*)d_in[2];
    const float* bout = (const float*)d_in[3];
    float* out = (float*)d_out;

    char* ws = (char*)d_ws;
    size_t off = 0;
    __bf16* xb    = (__bf16*)(ws + off); off += (size_t)BLc * Dc * 2;           // 16 MB
    __bf16* wqkvT = (__bf16*)(ws + off); off += (size_t)D3c * Dc * 2;           //  6 MB
    __bf16* woutT = (__bf16*)(ws + off); off += (size_t)Dc  * Dc * 2;           //  2 MB
    __bf16* qkv   = (__bf16*)(ws + off); off += (size_t)BLc * D3c * 2;          // 48 MB
    __bf16* vt    = (__bf16*)(ws + off); off += (size_t)Bc * Hc * HDc * Lc * 2; // 16 MB
    __bf16* Obuf  = (__bf16*)(ws + off); off += (size_t)BLc * Dc * 2;           // 16 MB

    {
        int n = BLc * Dc;
        cvt_f32_bf16_kernel<<<(n + 255) / 256, 256, 0, stream>>>(x, xb, n);
    }
    {
        int n = Dc * D3c;
        transpose_cvt_kernel<<<(n + 255) / 256, 256, 0, stream>>>(Wqkv, wqkvT, Dc, D3c);
    }
    {
        int n = Dc * Dc;
        transpose_cvt_kernel<<<(n + 255) / 256, 256, 0, stream>>>(Wout, woutT, Dc, Dc);
    }
    // QKV projection: [8192,1024] x [1024,3072] -> bf16 qkv
    gemm_bf16_kernel<true><<<dim3(D3c / 128, BLc / 128), 256, 0, stream>>>(
        xb, Dc, wqkvT, Dc, qkv, D3c, nullptr, Dc);
    // V transpose for PV B-matrix layout
    {
        int n = Bc * Hc * HDc * Lc;
        vtrans_kernel<<<(n + 255) / 256, 256, 0, stream>>>(qkv, vt);
    }
    // causal flash attention
    attn_kernel<<<dim3(Lc / 64, Bc * Hc), 128, 0, stream>>>(qkv, vt, Obuf);
    // output projection + bias: [8192,1024] x [1024,1024] -> fp32 out
    gemm_bf16_kernel<false><<<dim3(Dc / 128, BLc / 128), 256, 0, stream>>>(
        Obuf, Dc, woutT, Dc, out, Dc, bout, Dc);
}
